// GCN_G2_D2_86320252715553
// MI455X (gfx1250) — matrix operand order repaired and checked by reading, not compiled, not run on hardware
//
#include <hip/hip_runtime.h>

// ---------------------------------------------------------------------------
// GATv2 GNN forward for MI455X (gfx1250, wave32).
// Heavy math (x@W projections, 21 GFLOP) runs on the matrix pipe via
// v_wmma_f32_16x16x32_bf16. Each wave computes a 16x64 output strip so the
// A-fragment is reused across 4 WMMAs (load:wmma ratio 2.5:1 instead of 4:1,
// with 4 independent accumulator chains for latency hiding).
// Edge softmax/aggregation phases are L2-resident (xl+xr = 164MB < 192MB L2)
// gather/scatter with f32 atomics.
// ---------------------------------------------------------------------------

#define FIN  256
#define HEADS 8
#define CH   128
#define HC   1024     // HEADS*CH
#define DHID 128
#define NGRAPH 64

typedef __attribute__((ext_vector_type(16))) __bf16 v16bf;
typedef __attribute__((ext_vector_type(8)))  float  v8f;

__device__ __forceinline__ float leaky(float v, float s) {
    return v > 0.f ? v : s * v;
}

__device__ __forceinline__ void atomicMaxF32(float* addr, float v) {
    int* ai = (int*)addr;
    int old = __float_as_int(*addr);
    while (__int_as_float(old) < v) {
        int assumed = old;
        old = atomicCAS(ai, assumed, __float_as_int(v));
        if (old == assumed) break;
    }
}

// ---------------- generic fills / casts -----------------------------------

__global__ void k_fill_f32(float* __restrict__ p, float v, size_t n) {
    size_t i = (size_t)blockIdx.x * blockDim.x + threadIdx.x;
    if (i < n) p[i] = v;
}

__global__ void k_cvt_bf16(const float* __restrict__ in, __bf16* __restrict__ out, size_t n) {
    size_t i = (size_t)blockIdx.x * blockDim.x + threadIdx.x;
    if (i < n) out[i] = (__bf16)in[i];
}

// Pack W [K x Nf] (row major f32) into per-lane WMMA B-fragment order:
// layout [nTile16][kStep][lane][16 bf16]; B frag: lane holds column n=lane%16,
// element j holds K = kStep*32 + (lane/16)*8 + j for j<8, kStep*32 + 16 + (lane/16)*8 + (j-8) for j>=8.
__global__ void k_pack_w(const float* __restrict__ W, __bf16* __restrict__ Bp,
                         int K, int Nf) {
    int idx = blockIdx.x * blockDim.x + threadIdx.x;
    int total = Nf * K;
    if (idx >= total) return;
    int kSteps = K >> 5;
    int j    = idx & 15;
    int lane = (idx >> 4) & 31;
    int rest = idx >> 9;
    int ks   = rest % kSteps;
    int tn   = rest / kSteps;
    int half = lane >> 4;
    int k = ks * 32 + half * 8 + (j & 7) + ((j >> 3) << 4);
    int n = tn * 16 + (lane & 15);
    Bp[idx] = (__bf16)W[(size_t)k * Nf + n];
}

// ---------------- WMMA GEMM: C[M x Nf] = A[M x K](bf16) * Bpacked ---------
// grid = (ceil(M/16), Nf/256), block = 128 (4 waves);
// each wave computes a 16x64 strip = 4 accumulators sharing one A fragment.
__global__ void k_gemm_wmma_bf16(const __bf16* __restrict__ A,
                                 const __bf16* __restrict__ Bp,
                                 float* __restrict__ C,
                                 int M, int K, int Nf) {
    int lane = threadIdx.x & 31;
    int wave = threadIdx.x >> 5;
    int tm = blockIdx.x;
    int tn0 = (blockIdx.y * 4 + wave) * 4;   // first of 4 consecutive 16-col tiles
    int half = lane >> 4;
    int mrow = tm * 16 + (lane & 15);
    int mrowc = mrow < M ? mrow : M - 1;     // clamp loads; EXEC stays all-ones for WMMA
    int kSteps = K >> 5;

    v8f acc[4] = {};
    const __bf16* arow = A + (size_t)mrowc * K;
    const size_t tileStride = (size_t)kSteps * 512;   // elements per 16-col B tile
    const __bf16* b0 = Bp + (size_t)tn0 * tileStride;

    for (int ks = 0; ks < kSteps; ++ks) {
        // A fragment: 16-bit A 16x32 layout -> two contiguous 16B chunks per lane
        const __bf16* ap = arow + ks * 32 + half * 8;
        v16bf a;
        *((uint4*)&a)     = *(const uint4*)(ap);        // K = ks*32+half*8 .. +7
        *((uint4*)&a + 1) = *(const uint4*)(ap + 16);   // K = ks*32+16+half*8 .. +7
        // 4 B fragments: one contiguous 32B load each (packed order)
        size_t boff = ((size_t)ks * 32 + lane) * 16;
        v16bf bb[4];
#pragma unroll
        for (int t = 0; t < 4; ++t)
            bb[t] = *(const v16bf*)(b0 + (size_t)t * tileStride + boff);
#pragma unroll
        for (int t = 0; t < 4; ++t)
            acc[t] = __builtin_amdgcn_wmma_f32_16x16x32_bf16(
                         false, a, false, bb[t], (short)0, acc[t], false, false);
    }

    int mbase = tm * 16 + half * 8;
#pragma unroll
    for (int t = 0; t < 4; ++t) {
        int ncol = (tn0 + t) * 16 + (lane & 15);
        float* cp = C + (size_t)mbase * Nf + ncol;
#pragma unroll
        for (int r = 0; r < 8; ++r) {
            if (mbase + r < M) cp[(size_t)r * Nf] = acc[t][r];
        }
    }
}

// ---------------- conv1 edge phase ----------------------------------------

// One wave per edge. 4 lanes per head: head h = lane>>2, channel = (lane&3)+4i.
__global__ void k_edge_scores1(const float* __restrict__ xl, const float* __restrict__ xr,
                               const int* __restrict__ ei, const float* __restrict__ att,
                               float* __restrict__ e1, float* __restrict__ emax1,
                               int E, int Etot) {
    int eid = blockIdx.x * (blockDim.x >> 5) + (threadIdx.x >> 5);
    if (eid >= Etot) return;
    int lane = threadIdx.x & 31;
    int src = eid < E ? ei[eid]     : (eid - E);
    int dst = eid < E ? ei[E + eid] : (eid - E);
    int h  = lane >> 2;
    int cb = lane & 3;
    const float* pl = xl + (size_t)src * HC + h * CH;
    const float* pr = xr + (size_t)dst * HC + h * CH;
    const float* pa = att + h * CH;
    float acc = 0.f;
#pragma unroll 8
    for (int i = 0; i < 32; ++i) {
        int c = cb + (i << 2);
        acc += leaky(pl[c] + pr[c], 0.2f) * pa[c];
    }
    acc += __shfl_xor(acc, 1);
    acc += __shfl_xor(acc, 2);
    if (cb == 0) {
        e1[(size_t)eid * HEADS + h] = acc;
        atomicMaxF32(&emax1[(size_t)dst * HEADS + h], acc);
    }
}

__global__ void k_edge_exp1(const int* __restrict__ ei, float* __restrict__ e1,
                            const float* __restrict__ emax1, float* __restrict__ den1,
                            int E, int Etot) {
    size_t idx = (size_t)blockIdx.x * blockDim.x + threadIdx.x;
    if (idx >= (size_t)Etot * HEADS) return;
    int eid = (int)(idx >> 3);
    int h   = (int)(idx & 7);
    int dst = eid < E ? ei[E + eid] : (eid - E);
    float ee = __expf(e1[idx] - emax1[(size_t)dst * HEADS + h]);
    e1[idx] = ee;
    atomicAdd(&den1[(size_t)dst * HEADS + h], ee);
}

__global__ void k_aggregate1(const float* __restrict__ xl, const int* __restrict__ ei,
                             const float* __restrict__ e1, const float* __restrict__ den1,
                             float* __restrict__ out, int E, int Etot) {
    int eid = blockIdx.x * (blockDim.x >> 5) + (threadIdx.x >> 5);
    if (eid >= Etot) return;
    int lane = threadIdx.x & 31;
    int src = eid < E ? ei[eid]     : (eid - E);
    int dst = eid < E ? ei[E + eid] : (eid - E);
    int h  = lane >> 2;
    int cb = lane & 3;
    float alpha = e1[(size_t)eid * HEADS + h] /
                  (den1[(size_t)dst * HEADS + h] + 1e-16f);
    const float* pl = xl + (size_t)src * HC + h * CH;
    float* po = out + (size_t)dst * HC + h * CH;
#pragma unroll 8
    for (int i = 0; i < 32; ++i) {
        int c = cb + (i << 2);
        atomicAdd(&po[c], alpha * pl[c]);
    }
}

// ---------------- batchnorm 1 ---------------------------------------------

__global__ void k_bn1_stats(const float* __restrict__ Hm, float* __restrict__ mu,
                            float* __restrict__ rv, int N) {
    __shared__ float ss[256], ss2[256];
    int f = blockIdx.x;
    float s = 0.f, s2 = 0.f;
    for (int n = threadIdx.x; n < N; n += 256) {
        float v = Hm[(size_t)n * HC + f];
        s += v; s2 += v * v;
    }
    ss[threadIdx.x] = s; ss2[threadIdx.x] = s2;
    __syncthreads();
    for (int o = 128; o > 0; o >>= 1) {
        if (threadIdx.x < o) { ss[threadIdx.x] += ss[threadIdx.x + o];
                               ss2[threadIdx.x] += ss2[threadIdx.x + o]; }
        __syncthreads();
    }
    if (threadIdx.x == 0) {
        float m = ss[0] / N;
        float var = ss2[0] / N - m * m;
        mu[f] = m;
        rv[f] = rsqrtf(var + 1e-5f);
    }
}

__global__ void k_bn1_apply(float* __restrict__ Hm, const float* __restrict__ mu,
                            const float* __restrict__ rv, const float* __restrict__ g,
                            const float* __restrict__ be, size_t n) {
    size_t i = (size_t)blockIdx.x * blockDim.x + threadIdx.x;
    if (i >= n) return;
    int f = (int)(i & (HC - 1));
    float v = (Hm[i] - mu[f]) * rv[f] * g[f] + be[f];
    Hm[i] = leaky(v, 0.01f);
}

// ---------------- conv2 (heads=1, out=1) ----------------------------------

__global__ void k_conv2_proj(const float* __restrict__ Hm, const float* __restrict__ w2l,
                             const float* __restrict__ w2r, float* __restrict__ xl2,
                             float* __restrict__ xr2, int N) {
    int n = blockIdx.x * (blockDim.x >> 5) + (threadIdx.x >> 5);
    if (n >= N) return;
    int lane = threadIdx.x & 31;
    const float* hp = Hm + (size_t)n * HC;
    float al = 0.f, ar = 0.f;
    for (int c = lane; c < HC; c += 32) {
        float v = hp[c];
        al += v * w2l[c];
        ar += v * w2r[c];
    }
    for (int o = 16; o > 0; o >>= 1) {
        al += __shfl_down(al, o);
        ar += __shfl_down(ar, o);
    }
    if (lane == 0) { xl2[n] = al; xr2[n] = ar; }
}

__global__ void k_edge_scores2(const float* __restrict__ xl2, const float* __restrict__ xr2,
                               const int* __restrict__ ei, const float* __restrict__ att2,
                               float* __restrict__ e2, float* __restrict__ emax2,
                               int E, int Etot) {
    int eid = blockIdx.x * blockDim.x + threadIdx.x;
    if (eid >= Etot) return;
    int src = eid < E ? ei[eid]     : (eid - E);
    int dst = eid < E ? ei[E + eid] : (eid - E);
    float s = leaky(xl2[src] + xr2[dst], 0.2f) * att2[0];
    e2[eid] = s;
    atomicMaxF32(&emax2[dst], s);
}

__global__ void k_edge_exp2(const int* __restrict__ ei, float* __restrict__ e2,
                            const float* __restrict__ emax2, float* __restrict__ den2,
                            int E, int Etot) {
    int eid = blockIdx.x * blockDim.x + threadIdx.x;
    if (eid >= Etot) return;
    int dst = eid < E ? ei[E + eid] : (eid - E);
    float ee = __expf(e2[eid] - emax2[dst]);
    e2[eid] = ee;
    atomicAdd(&den2[dst], ee);
}

__global__ void k_aggregate2(const float* __restrict__ xl2, const int* __restrict__ ei,
                             const float* __restrict__ e2, const float* __restrict__ den2,
                             float* __restrict__ h2, int E, int Etot) {
    int eid = blockIdx.x * blockDim.x + threadIdx.x;
    if (eid >= Etot) return;
    int src = eid < E ? ei[eid]     : (eid - E);
    int dst = eid < E ? ei[E + eid] : (eid - E);
    atomicAdd(&h2[dst], (e2[eid] / (den2[dst] + 1e-16f)) * xl2[src]);
}

// ---------------- batchnorm 2 + pooling -----------------------------------

__global__ void k_bn2_stats(const float* __restrict__ h2, float* __restrict__ st, int N) {
    __shared__ float ss[1024], ss2[1024];
    float s = 0.f, s2 = 0.f;
    for (int n = threadIdx.x; n < N; n += 1024) {
        float v = h2[n];
        s += v; s2 += v * v;
    }
    ss[threadIdx.x] = s; ss2[threadIdx.x] = s2;
    __syncthreads();
    for (int o = 512; o > 0; o >>= 1) {
        if (threadIdx.x < o) { ss[threadIdx.x] += ss[threadIdx.x + o];
                               ss2[threadIdx.x] += ss2[threadIdx.x + o]; }
        __syncthreads();
    }
    if (threadIdx.x == 0) {
        float m = ss[0] / N;
        float var = ss2[0] / N - m * m;
        st[0] = m;
        st[1] = rsqrtf(var + 1e-5f);
    }
}

__global__ void k_bn2_pool(const float* __restrict__ h2, const float* __restrict__ st,
                           const float* __restrict__ g2, const float* __restrict__ be2,
                           const int* __restrict__ batch, float* __restrict__ pool,
                           float* __restrict__ cnt, int N) {
    int n = blockIdx.x * blockDim.x + threadIdx.x;
    if (n >= N) return;
    float v = (h2[n] - st[0]) * st[1] * g2[0] + be2[0];
    v = leaky(v, 0.01f);
    int gidx = batch[n];
    atomicAdd(&pool[gidx], v);
    atomicAdd(&cnt[gidx], 1.0f);
}

// ---------------- MLP head -------------------------------------------------

__global__ void k_mlp(const float* __restrict__ pool, const float* __restrict__ cnt,
                      const float* __restrict__ lw1, const float* __restrict__ lb1,
                      const float* __restrict__ lw2, const float* __restrict__ lb2,
                      float* __restrict__ out) {
    __shared__ float ss[DHID];
    int gidx = blockIdx.x;
    int d = threadIdx.x;
    float p = pool[gidx] / fmaxf(cnt[gidx], 1.0f);
    float hd = leaky(p * lw1[d] + lb1[d], 0.01f);
    ss[d] = hd * lw2[d];
    __syncthreads();
    for (int o = DHID / 2; o > 0; o >>= 1) {
        if (d < o) ss[d] += ss[d + o];
        __syncthreads();
    }
    if (d == 0) out[gidx] = ss[0] + lb2[0];
}

// ---------------------------------------------------------------------------

extern "C" void kernel_launch(void* const* d_in, const int* in_sizes, int n_in,
                              void* d_out, int out_size, void* d_ws, size_t ws_size,
                              hipStream_t stream) {
    const float* x    = (const float*)d_in[0];
    const int*   ei   = (const int*)d_in[1];
    const int*   batch= (const int*)d_in[2];
    const float* w1l  = (const float*)d_in[3];
    const float* w1r  = (const float*)d_in[4];
    const float* att1 = (const float*)d_in[5];
    const float* g1   = (const float*)d_in[7];
    const float* be1  = (const float*)d_in[8];
    const float* w2l  = (const float*)d_in[9];
    const float* w2r  = (const float*)d_in[10];
    const float* att2 = (const float*)d_in[11];
    const float* g2   = (const float*)d_in[13];
    const float* be2  = (const float*)d_in[14];
    const float* lw1  = (const float*)d_in[15];
    const float* lb1  = (const float*)d_in[16];
    const float* lw2  = (const float*)d_in[17];
    const float* lb2  = (const float*)d_in[18];

    const int N    = in_sizes[0] / FIN;
    const int E    = in_sizes[1] / 2;
    const int Etot = E + N;

    // --- workspace layout ---
    char* ws = (char*)d_ws;
    size_t off = 0;
    auto alloc = [&](size_t bytes) -> char* {
        char* p = ws + off;
        off += (bytes + 255) & ~(size_t)255;
        return p;
    };
    __bf16* xb    = (__bf16*)alloc((size_t)N * FIN * 2);
    __bf16* w1lp  = (__bf16*)alloc((size_t)FIN * HC * 2);
    __bf16* w1rp  = (__bf16*)alloc((size_t)FIN * HC * 2);
    float*  xl    = (float*)alloc((size_t)N * HC * 4);
    float*  xr    = (float*)alloc((size_t)N * HC * 4);   // reused as conv1 output
    float*  e1    = (float*)alloc((size_t)Etot * HEADS * 4);
    float*  emax1 = (float*)alloc((size_t)N * HEADS * 4);
    float*  den1  = (float*)alloc((size_t)N * HEADS * 4);
    float*  mu1   = (float*)alloc(HC * 4);
    float*  rv1   = (float*)alloc(HC * 4);
    float*  xl2   = (float*)alloc((size_t)N * 4);
    float*  xr2   = (float*)alloc((size_t)N * 4);
    float*  e2    = (float*)alloc((size_t)Etot * 4);
    float*  emax2 = (float*)alloc((size_t)N * 4);
    float*  den2  = (float*)alloc((size_t)N * 4);
    float*  h2    = (float*)alloc((size_t)N * 4);
    float*  st2   = (float*)alloc(2 * 4);
    float*  pool  = (float*)alloc(NGRAPH * 4);
    float*  cnt   = (float*)alloc(NGRAPH * 4);

    const size_t nXF = (size_t)N * FIN;
    const size_t nH  = (size_t)N * HC;

    // 1) cast x to bf16; pack weights into WMMA-B fragment order
    k_cvt_bf16<<<(int)((nXF + 255) / 256), 256, 0, stream>>>(x, xb, nXF);
    k_pack_w<<<(FIN * HC + 255) / 256, 256, 0, stream>>>(w1l, w1lp, FIN, HC);
    k_pack_w<<<(FIN * HC + 255) / 256, 256, 0, stream>>>(w1r, w1rp, FIN, HC);

    // 2) WMMA GEMMs: xl = x@w1_l, xr = x@w1_r  (M x 1024, K=256)
    dim3 gg((N + 15) / 16, HC / 256);
    k_gemm_wmma_bf16<<<gg, 128, 0, stream>>>(xb, w1lp, xl, N, FIN, HC);
    k_gemm_wmma_bf16<<<gg, 128, 0, stream>>>(xb, w1rp, xr, N, FIN, HC);

    // 3) conv1 edge softmax + aggregation
    size_t nNH = (size_t)N * HEADS;
    k_fill_f32<<<(int)((nNH + 255) / 256), 256, 0, stream>>>(emax1, -1e30f, nNH);
    k_fill_f32<<<(int)((nNH + 255) / 256), 256, 0, stream>>>(den1, 0.f, nNH);
    int edgeWaveBlocks = (Etot + 7) / 8;
    k_edge_scores1<<<edgeWaveBlocks, 256, 0, stream>>>(xl, xr, ei, att1, e1, emax1, E, Etot);
    // xr no longer needed as projection -> becomes conv1 output accumulator
    k_fill_f32<<<(int)((nH + 255) / 256), 256, 0, stream>>>(xr, 0.f, nH);
    size_t nEH = (size_t)Etot * HEADS;
    k_edge_exp1<<<(int)((nEH + 255) / 256), 256, 0, stream>>>(ei, e1, emax1, den1, E, Etot);
    k_aggregate1<<<edgeWaveBlocks, 256, 0, stream>>>(xl, ei, e1, den1, xr, E, Etot);

    // 4) batchnorm1 + leaky relu (bias b1 cancels inside BN)
    k_bn1_stats<<<HC, 256, 0, stream>>>(xr, mu1, rv1, N);
    k_bn1_apply<<<(int)((nH + 255) / 256), 256, 0, stream>>>(xr, mu1, rv1, g1, be1, nH);

    // 5) conv2 projections (dot with 1024-vectors)
    k_conv2_proj<<<(N + 7) / 8, 256, 0, stream>>>(xr, w2l, w2r, xl2, xr2, N);

    // 6) conv2 edge softmax + aggregation (scalar per edge)
    k_fill_f32<<<(N + 255) / 256, 256, 0, stream>>>(emax2, -1e30f, (size_t)N);
    k_fill_f32<<<(N + 255) / 256, 256, 0, stream>>>(den2, 0.f, (size_t)N);
    k_fill_f32<<<(N + 255) / 256, 256, 0, stream>>>(h2, 0.f, (size_t)N);
    k_fill_f32<<<1, 256, 0, stream>>>(pool, 0.f, (size_t)NGRAPH);
    k_fill_f32<<<1, 256, 0, stream>>>(cnt, 0.f, (size_t)NGRAPH);
    int edgeBlocks = (Etot + 255) / 256;
    k_edge_scores2<<<edgeBlocks, 256, 0, stream>>>(xl2, xr2, ei, att2, e2, emax2, E, Etot);
    k_edge_exp2<<<edgeBlocks, 256, 0, stream>>>(ei, e2, emax2, den2, E, Etot);
    k_aggregate2<<<edgeBlocks, 256, 0, stream>>>(xl2, ei, e2, den2, h2, E, Etot);

    // 7) batchnorm2 + leaky + global mean pool
    k_bn2_stats<<<1, 1024, 0, stream>>>(h2, st2, N);
    k_bn2_pool<<<(N + 255) / 256, 256, 0, stream>>>(h2, st2, g2, be2, batch, pool, cnt, N);

    // 8) MLP head -> out [64, 1]
    k_mlp<<<NGRAPH, DHID, 0, stream>>>(pool, cnt, lw1, lb1, lw2, lb2, (float*)d_out);
}
